// PWAM_24404004176523
// MI455X (gfx1250) — matrix operand order repaired and checked by reading, not compiled, not run on hardware
//
#include <hip/hip_runtime.h>

typedef __attribute__((ext_vector_type(16))) _Float16 v16h;
typedef __attribute__((ext_vector_type(8)))  _Float16 v8h;
typedef __attribute__((ext_vector_type(8)))  float    v8f;

#define WMMA_F16(A, B, C) \
  __builtin_amdgcn_wmma_f32_16x16x32_f16(false, (A), false, (B), (short)0, (C), false, false)

#define HW    16384
#define BATCH 8
#define DCH   512
#define NLP   32   // padded language tokens (real = 20)
#define LDA   40   // LDS row stride in halfs

// Branch-free GELU (tanh form) using hardware v_exp_f32; overflow-safe.
__device__ __forceinline__ float gelu_fast(float x) {
  float u = x * (0.7978845608028654f + 0.03567740813636141f * x * x);
  float e = __expf(2.0f * u);
  float th = 1.0f - 2.0f / (e + 1.0f);   // tanh(u), safe as e -> inf or 0
  return 0.5f * x * (1.0f + th);
}

// ---- CDNA5 async global->LDS copy (ASYNCcnt) -------------------------------
__device__ __forceinline__ void async_ld_b128(void* lds_dst, const void* gsrc) {
  unsigned int l = (unsigned int)(unsigned long long)lds_dst;  // low 32 = LDS offset
  asm volatile("global_load_async_to_lds_b128 %0, %1, off"
               :: "v"(l), "v"(gsrc) : "memory");
}
__device__ __forceinline__ void wait_async0() {
  asm volatile("s_wait_asynccnt 0x0" ::: "memory");
}

// ---- WMMA fragment helpers -------------------------------------------------
// A (16x32 f16): lanes 0-15 hold K 0..7 & 16..23, lanes 16-31 hold K 8..15 & 24..31
__device__ __forceinline__ v16h afrag_lds(const _Float16* As, int row, int half16) {
  const _Float16* p = As + row * LDA + half16 * 8;
  v8h lo = *(const v8h*)p;
  v8h hi = *(const v8h*)(p + 16);
  v16h a;
#pragma unroll
  for (int i = 0; i < 8; ++i) { a[i] = lo[i]; a[i + 8] = hi[i]; }
  return a;
}
// B (32x16 f16): lanes 0-15 = column n, K 0..7 & 16..23; lanes 16-31 = column n, K 8..15 & 24..31
__device__ __forceinline__ v16h bfrag_lds(const _Float16* Bs, int col, int half16) {
  const _Float16* p = Bs + col * LDA + half16 * 8;
  v8h lo = *(const v8h*)p;
  v8h hi = *(const v8h*)(p + 16);
  v16h b;
#pragma unroll
  for (int i = 0; i < 8; ++i) { b[i] = lo[i]; b[i + 8] = hi[i]; }
  return b;
}
__device__ __forceinline__ v16h load16g(const _Float16* p) {
  v8h lo = *(const v8h*)p;
  v8h hi = *(const v8h*)(p + 8);
  v16h r;
#pragma unroll
  for (int i = 0; i < 8; ++i) { r[i] = lo[i]; r[i + 8] = hi[i]; }
  return r;
}

// ---- K1: dual GEMM  vis = gelu(x@Wv+bv) -> f16 ; q = x@Wq+bq -> f16 --------
// Fused instance-norm partial stats for the q path.
// grid (8, 1024): x = wsel*4 + ntile, y = M tile. 256 threads = 8 waves.
__global__ __launch_bounds__(256) void k_gemm_x(const float* __restrict__ X,
                                                const float* __restrict__ Wv,
                                                const float* __restrict__ bv,
                                                const float* __restrict__ Wq,
                                                const float* __restrict__ bq,
                                                _Float16* __restrict__ visb,
                                                _Float16* __restrict__ qb,
                                                float* __restrict__ pS,
                                                float* __restrict__ pQ) {
  __shared__ _Float16 As[128 * LDA];
  __shared__ _Float16 Bs[128 * LDA];
  __shared__ float sRed[2][128];
  __shared__ float qRed[2][128];
  const int t = threadIdx.x, lane = t & 31, wid = t >> 5;
  const int wm = wid & 1, wn = wid >> 1;
  const int r = lane & 15, half16 = lane >> 4;
  const int wsel = blockIdx.x >> 2;
  const int nbase = (blockIdx.x & 3) * 128;
  const int mbase = blockIdx.y * 128;
  const float* W = wsel ? Wq : Wv;
  const float* bias = wsel ? bq : bv;
  _Float16* Out = wsel ? qb : visb;

  v8f acc[4][2];
#pragma unroll
  for (int i = 0; i < 4; ++i) { acc[i][0] = (v8f){}; acc[i][1] = (v8f){}; }

  const int arow = t >> 1, acg = (t & 1) * 16;
  const int bkk = t >> 5, bnn = (t & 31) * 4;

  for (int kc = 0; kc < 512; kc += 32) {
    __syncthreads();
    // A tile: fp32 -> f16
    {
      const float* src = X + (size_t)(mbase + arow) * DCH + kc + acg;
      _Float16* dst = As + arow * LDA + acg;
#pragma unroll
      for (int u = 0; u < 4; ++u) {
        float4 v = *(const float4*)(src + u * 4);
        dst[u * 4 + 0] = (_Float16)v.x;
        dst[u * 4 + 1] = (_Float16)v.y;
        dst[u * 4 + 2] = (_Float16)v.z;
        dst[u * 4 + 3] = (_Float16)v.w;
      }
    }
    // B tile: W[k][n] -> Bs[n][k] (transposed, f16)
#pragma unroll
    for (int p = 0; p < 4; ++p) {
      int k = bkk + p * 8;
      float4 w = *(const float4*)(W + (size_t)(kc + k) * DCH + nbase + bnn);
      Bs[(bnn + 0) * LDA + k] = (_Float16)w.x;
      Bs[(bnn + 1) * LDA + k] = (_Float16)w.y;
      Bs[(bnn + 2) * LDA + k] = (_Float16)w.z;
      Bs[(bnn + 3) * LDA + k] = (_Float16)w.w;
    }
    __syncthreads();
    v16h b0 = bfrag_lds(Bs, wn * 32 + r, half16);
    v16h b1 = bfrag_lds(Bs, wn * 32 + 16 + r, half16);
#pragma unroll
    for (int i = 0; i < 4; ++i) {
      v16h a = afrag_lds(As, wm * 64 + i * 16 + r, half16);
      acc[i][0] = WMMA_F16(a, b0, acc[i][0]);
      acc[i][1] = WMMA_F16(a, b1, acc[i][1]);
    }
  }
  float sacc[2] = {0.f, 0.f}, qacc[2] = {0.f, 0.f};
#pragma unroll
  for (int i = 0; i < 4; ++i)
#pragma unroll
    for (int j = 0; j < 2; ++j) {
      int ncol = nbase + wn * 32 + j * 16 + r;
      float bb = bias[ncol];
#pragma unroll
      for (int e = 0; e < 8; ++e) {
        int mrow = mbase + wm * 64 + i * 16 + half16 * 8 + e;
        float v = acc[i][j][e] + bb;
        if (wsel == 0) {
          Out[(size_t)mrow * DCH + ncol] = (_Float16)gelu_fast(v);
        } else {
          Out[(size_t)mrow * DCH + ncol] = (_Float16)v;
          sacc[j] += v;
          qacc[j] += v * v;
        }
      }
    }
  if (wsel == 1) {  // fused instance-norm partials (block-uniform branch)
#pragma unroll
    for (int j = 0; j < 2; ++j) {
      float s = sacc[j] + __shfl_xor(sacc[j], 16, 32);
      float qq = qacc[j] + __shfl_xor(qacc[j], 16, 32);
      if (half16 == 0) {
        sRed[wm][wn * 32 + j * 16 + r] = s;
        qRed[wm][wn * 32 + j * 16 + r] = qq;
      }
    }
    __syncthreads();
    if (t < 128) {
      pS[(size_t)blockIdx.y * DCH + nbase + t] = sRed[0][t] + sRed[1][t];
      pQ[(size_t)blockIdx.y * DCH + nbase + t] = qRed[0][t] + qRed[1][t];
    }
  }
}

// ---- GEMM  out_f16 = A_f16 @ Ww + bw, fused stats; async A-tile copies -----
// grid (4, 1024)
__global__ __launch_bounds__(256) void k_gemm_w(const _Float16* __restrict__ Ain,
                                                const float* __restrict__ W,
                                                const float* __restrict__ bias,
                                                _Float16* __restrict__ Out,
                                                float* __restrict__ pS,
                                                float* __restrict__ pQ) {
  __shared__ _Float16 As[128 * LDA];
  __shared__ _Float16 Bs[128 * LDA];
  __shared__ float sRed[2][128];
  __shared__ float qRed[2][128];
  const int t = threadIdx.x, lane = t & 31, wid = t >> 5;
  const int wm = wid & 1, wn = wid >> 1;
  const int r = lane & 15, half16 = lane >> 4;
  const int nbase = blockIdx.x * 128;
  const int mbase = blockIdx.y * 128;

  v8f acc[4][2];
#pragma unroll
  for (int i = 0; i < 4; ++i) { acc[i][0] = (v8f){}; acc[i][1] = (v8f){}; }

  const int arow = t >> 1, acg = (t & 1) * 16;
  const int bkk = t >> 5, bnn = (t & 31) * 4;

  for (int kc = 0; kc < 512; kc += 32) {
    __syncthreads();
    // A tile: pure f16 copy -> CDNA5 async global->LDS (ASYNCcnt)
    {
      const _Float16* src = Ain + (size_t)(mbase + arow) * DCH + kc + acg;
      async_ld_b128(As + arow * LDA + acg, src);
      async_ld_b128(As + arow * LDA + acg + 8, src + 8);
    }
    // B tile: W[k][n] -> Bs[n][k] (transposed, f16)
#pragma unroll
    for (int p = 0; p < 4; ++p) {
      int k = bkk + p * 8;
      float4 w = *(const float4*)(W + (size_t)(kc + k) * DCH + nbase + bnn);
      Bs[(bnn + 0) * LDA + k] = (_Float16)w.x;
      Bs[(bnn + 1) * LDA + k] = (_Float16)w.y;
      Bs[(bnn + 2) * LDA + k] = (_Float16)w.z;
      Bs[(bnn + 3) * LDA + k] = (_Float16)w.w;
    }
    wait_async0();
    __syncthreads();
    v16h b0 = bfrag_lds(Bs, wn * 32 + r, half16);
    v16h b1 = bfrag_lds(Bs, wn * 32 + 16 + r, half16);
#pragma unroll
    for (int i = 0; i < 4; ++i) {
      v16h a = afrag_lds(As, wm * 64 + i * 16 + r, half16);
      acc[i][0] = WMMA_F16(a, b0, acc[i][0]);
      acc[i][1] = WMMA_F16(a, b1, acc[i][1]);
    }
  }
  float sacc[2] = {0.f, 0.f}, qacc[2] = {0.f, 0.f};
#pragma unroll
  for (int i = 0; i < 4; ++i)
#pragma unroll
    for (int j = 0; j < 2; ++j) {
      int ncol = nbase + wn * 32 + j * 16 + r;
      float bb = bias[ncol];
#pragma unroll
      for (int e = 0; e < 8; ++e) {
        int mrow = mbase + wm * 64 + i * 16 + half16 * 8 + e;
        float v = acc[i][j][e] + bb;
        Out[(size_t)mrow * DCH + ncol] = (_Float16)v;
        sacc[j] += v;
        qacc[j] += v * v;
      }
    }
#pragma unroll
  for (int j = 0; j < 2; ++j) {
    float s = sacc[j] + __shfl_xor(sacc[j], 16, 32);
    float qq = qacc[j] + __shfl_xor(qacc[j], 16, 32);
    if (half16 == 0) {
      sRed[wm][wn * 32 + j * 16 + r] = s;
      qRed[wm][wn * 32 + j * 16 + r] = qq;
    }
  }
  __syncthreads();
  if (t < 128) {
    pS[(size_t)blockIdx.y * DCH + nbase + t] = sRed[0][t] + sRed[1][t];
    pQ[(size_t)blockIdx.y * DCH + nbase + t] = qRed[0][t] + qRed[1][t];
  }
}

// ---- K_final: out = gelu( (vis * inorm(wraw)) @ Wm + bm ) ------------------
__global__ __launch_bounds__(256) void k_gemm_mm(const _Float16* __restrict__ vis,
                                                 const _Float16* __restrict__ wr,
                                                 const float* __restrict__ mean2,
                                                 const float* __restrict__ inv2,
                                                 const float* __restrict__ W,
                                                 const float* __restrict__ bias,
                                                 float* __restrict__ Out) {
  __shared__ _Float16 As[128 * LDA];
  __shared__ _Float16 Bs[128 * LDA];
  __shared__ float meanS[512];
  __shared__ float invS[512];
  const int t = threadIdx.x, lane = t & 31, wid = t >> 5;
  const int wm = wid & 1, wn = wid >> 1;
  const int r = lane & 15, half16 = lane >> 4;
  const int nbase = blockIdx.x * 128;
  const int mbase = blockIdx.y * 128;
  const int b = mbase / HW;
  for (int i = t; i < 512; i += 256) {
    meanS[i] = mean2[b * 512 + i];
    invS[i] = inv2[b * 512 + i];
  }

  v8f acc[4][2];
#pragma unroll
  for (int i = 0; i < 4; ++i) { acc[i][0] = (v8f){}; acc[i][1] = (v8f){}; }

  const int arow = t >> 1, acg = (t & 1) * 16;
  const int bkk = t >> 5, bnn = (t & 31) * 4;

  for (int kc = 0; kc < 512; kc += 32) {
    __syncthreads();
    {
      size_t off = (size_t)(mbase + arow) * DCH + kc + acg;
      v8h v0 = *(const v8h*)(vis + off);
      v8h v1 = *(const v8h*)(vis + off + 8);
      v8h w0 = *(const v8h*)(wr + off);
      v8h w1 = *(const v8h*)(wr + off + 8);
      _Float16* dst = As + arow * LDA + acg;
#pragma unroll
      for (int i = 0; i < 8; ++i) {
        int c = kc + acg + i;
        dst[i] = (_Float16)((float)v0[i] * (((float)w0[i] - meanS[c]) * invS[c]));
        dst[i + 8] = (_Float16)((float)v1[i] * (((float)w1[i] - meanS[c + 8]) * invS[c + 8]));
      }
    }
#pragma unroll
    for (int p = 0; p < 4; ++p) {
      int k = bkk + p * 8;
      float4 w = *(const float4*)(W + (size_t)(kc + k) * DCH + nbase + bnn);
      Bs[(bnn + 0) * LDA + k] = (_Float16)w.x;
      Bs[(bnn + 1) * LDA + k] = (_Float16)w.y;
      Bs[(bnn + 2) * LDA + k] = (_Float16)w.z;
      Bs[(bnn + 3) * LDA + k] = (_Float16)w.w;
    }
    __syncthreads();
    v16h b0 = bfrag_lds(Bs, wn * 32 + r, half16);
    v16h b1 = bfrag_lds(Bs, wn * 32 + 16 + r, half16);
#pragma unroll
    for (int i = 0; i < 4; ++i) {
      v16h a = afrag_lds(As, wm * 64 + i * 16 + r, half16);
      acc[i][0] = WMMA_F16(a, b0, acc[i][0]);
      acc[i][1] = WMMA_F16(a, b1, acc[i][1]);
    }
  }
#pragma unroll
  for (int i = 0; i < 4; ++i)
#pragma unroll
    for (int j = 0; j < 2; ++j) {
      int ncol = nbase + wn * 32 + j * 16 + r;
      float bb = bias[ncol];
#pragma unroll
      for (int e = 0; e < 8; ++e) {
        int mrow = mbase + wm * 64 + i * 16 + half16 * 8 + e;
        Out[(size_t)mrow * DCH + ncol] = gelu_fast(acc[i][j][e] + bb);
      }
    }
}

// ---- finalize stats: fold 128 deterministic per-Mtile partials -------------
__global__ __launch_bounds__(256) void k_finstats(const float* __restrict__ pS,
                                                  const float* __restrict__ pQ,
                                                  float* __restrict__ meanA,
                                                  float* __restrict__ invA) {
  int i = blockIdx.x * 256 + threadIdx.x;  // 0..4095 = (b, c)
  int b = i >> 9, c = i & 511;
  float s = 0.f, q = 0.f;
  for (int m = 0; m < 128; ++m) {
    size_t idx = ((size_t)(b * 128 + m)) * DCH + c;
    s += pS[idx];
    q += pQ[idx];
  }
  float mn = s * (1.0f / 16384.0f);
  float var = q * (1.0f / 16384.0f) - mn * mn;
  meanA[i] = mn;
  invA[i] = rsqrtf(var + 1e-5f);
}

// ---- K/V projection (tiny): k token-major padded to 32, v feature-major ----
__global__ __launch_bounds__(256) void k_kv(const float* __restrict__ l,
                                            const float* __restrict__ Wk,
                                            const float* __restrict__ bk,
                                            const float* __restrict__ Wval,
                                            const float* __restrict__ bval,
                                            _Float16* __restrict__ kbuf,
                                            _Float16* __restrict__ vT) {
  __shared__ float lrow[768];
  const int b = blockIdx.x >> 5, n = blockIdx.x & 31, t = threadIdx.x;
  const int c0 = t, c1 = t + 256;
  if (n >= 20) {  // zero padding rows (block-uniform branch)
    kbuf[(size_t)(b * NLP + n) * DCH + c0] = (_Float16)0.f;
    kbuf[(size_t)(b * NLP + n) * DCH + c1] = (_Float16)0.f;
    vT[((size_t)b * DCH + c0) * NLP + n] = (_Float16)0.f;
    vT[((size_t)b * DCH + c1) * NLP + n] = (_Float16)0.f;
    return;
  }
  for (int i = t; i < 768; i += 256) lrow[i] = l[(size_t)(b * 20 + n) * 768 + i];
  __syncthreads();
  float aK0 = bk[c0], aK1 = bk[c1], aV0 = bval[c0], aV1 = bval[c1];
  for (int j = 0; j < 768; ++j) {
    float lv = lrow[j];
    aK0 = fmaf(lv, Wk[(size_t)j * DCH + c0], aK0);
    aK1 = fmaf(lv, Wk[(size_t)j * DCH + c1], aK1);
    aV0 = fmaf(lv, Wval[(size_t)j * DCH + c0], aV0);
    aV1 = fmaf(lv, Wval[(size_t)j * DCH + c1], aV1);
  }
  kbuf[(size_t)(b * NLP + n) * DCH + c0] = (_Float16)aK0;
  kbuf[(size_t)(b * NLP + n) * DCH + c1] = (_Float16)aK1;
  vT[((size_t)b * DCH + c0) * NLP + n] = (_Float16)aV0;
  vT[((size_t)b * DCH + c1) * NLP + n] = (_Float16)aV1;
}

// ---- Attention: per wave 16 queries x 8 heads, WMMA qk / softmax / pv ------
// grid (128, 8): x = query chunk of 128, y = batch. 256 threads = 8 waves.
__global__ __launch_bounds__(256) void k_attn(const _Float16* __restrict__ qb,
                                              const float* __restrict__ mean1,
                                              const float* __restrict__ inv1,
                                              const _Float16* __restrict__ kb,
                                              const _Float16* __restrict__ vT,
                                              _Float16* __restrict__ ao) {
  __shared__ float meanS[512];
  __shared__ float invS[512];
  __shared__ _Float16 Ps[8][512];  // per-wave 16x32 prob scratch
  const int t = threadIdx.x, lane = t & 31, wid = t >> 5;
  const int r = lane & 15, half16 = lane >> 4;
  const int b = blockIdx.y;
  for (int i = t; i < 512; i += 256) {
    meanS[i] = mean1[b * 512 + i];
    invS[i] = inv1[b * 512 + i];
  }
  __syncthreads();
  const int qbase = blockIdx.x * 128 + wid * 16;
  const size_t qrow = ((size_t)b * HW + qbase + r) * DCH;
  const float scale = 0.044194173824159216f;  // 512^-0.5
  _Float16* ps = &Ps[wid][0];

  for (int h = 0; h < 8; ++h) {
    const int cb = h * 64;
    v8f s0 = (v8f){}, s1 = (v8f){};
#pragma unroll
    for (int kc2 = 0; kc2 < 2; ++kc2) {
      const int cbase = cb + kc2 * 32;
      const int kh = half16 * 16;
      v8h lo = *(const v8h*)(qb + qrow + cbase + kh);
      v8h hi = *(const v8h*)(qb + qrow + cbase + kh + 8);
      v16h a;
#pragma unroll
      for (int i = 0; i < 8; ++i) {
        int c = cbase + kh + i;
        a[i] = (_Float16)(((float)lo[i] - meanS[c]) * invS[c]);
        a[i + 8] = (_Float16)(((float)hi[i] - meanS[c + 8]) * invS[c + 8]);
      }
      const int ko = half16 * 16;
      v16h bk0 = load16g(kb + ((size_t)(b * NLP) + r) * DCH + cbase + ko);
      v16h bk1 = load16g(kb + ((size_t)(b * NLP) + 16 + r) * DCH + cbase + ko);
      s0 = WMMA_F16(a, bk0, s0);
      s1 = WMMA_F16(a, bk1, s1);
    }
    // scale + token mask + rowwise softmax (rows split across 16-lane groups)
    float p0[8], p1[8];
#pragma unroll
    for (int e = 0; e < 8; ++e) {
      float v0 = s0[e] * scale;
      float v1 = (r >= 4) ? -1e30f : s1[e] * scale;  // tokens 20..31 masked
      float mx = fmaxf(v0, v1);
#pragma unroll
      for (int o = 8; o >= 1; o >>= 1) mx = fmaxf(mx, __shfl_xor(mx, o, 16));
      float e0 = __expf(v0 - mx);
      float e1 = __expf(v1 - mx);
      float sm = e0 + e1;
#pragma unroll
      for (int o = 8; o >= 1; o >>= 1) sm += __shfl_xor(sm, o, 16);
      float is = 1.0f / sm;
      p0[e] = e0 * is;
      p1[e] = e1 * is;
    }
    // transpose probs C-layout -> A-layout via wave-private LDS
    const int pr = half16 * 8;
#pragma unroll
    for (int e = 0; e < 8; ++e) {
      ps[(pr + e) * 32 + r] = (_Float16)p0[e];
      ps[(pr + e) * 32 + 16 + r] = (_Float16)p1[e];
    }
    const int kh2 = half16 * 16;
    v8h plo = *(const v8h*)(ps + r * 32 + kh2);
    v8h phi = *(const v8h*)(ps + r * 32 + kh2 + 8);
    v16h ap;
#pragma unroll
    for (int i = 0; i < 8; ++i) { ap[i] = plo[i]; ap[i + 8] = phi[i]; }
    const int ko2 = half16 * 16;
#pragma unroll
    for (int j2 = 0; j2 < 4; ++j2) {
      const int f = cb + j2 * 16 + r;
      v16h bv_ = load16g(vT + ((size_t)b * DCH + f) * NLP + ko2);
      v8f zero = (v8f){};
      v8f o = WMMA_F16(ap, bv_, zero);
#pragma unroll
      for (int e = 0; e < 8; ++e) {
        int mrow = qbase + half16 * 8 + e;
        ao[((size_t)b * HW + mrow) * DCH + f] = (_Float16)o[e];
      }
    }
  }
}

extern "C" void kernel_launch(void* const* d_in, const int* in_sizes, int n_in,
                              void* d_out, int out_size, void* d_ws, size_t ws_size,
                              hipStream_t stream) {
  (void)in_sizes; (void)n_in; (void)out_size; (void)ws_size;
  const float* x    = (const float*)d_in[0];
  const float* l    = (const float*)d_in[1];
  // d_in[2] = l_mask: all-ones and unused by the reference
  const float* Wv   = (const float*)d_in[3];
  const float* bv   = (const float*)d_in[4];
  const float* Wq   = (const float*)d_in[5];
  const float* bq   = (const float*)d_in[6];
  const float* Wk   = (const float*)d_in[7];
  const float* bk   = (const float*)d_in[8];
  const float* Wval = (const float*)d_in[9];
  const float* bval = (const float*)d_in[10];
  const float* Ww   = (const float*)d_in[11];
  const float* bw   = (const float*)d_in[12];
  const float* Wm   = (const float*)d_in[13];
  const float* bm   = (const float*)d_in[14];
  float* out = (float*)d_out;

  const size_t MTOT = (size_t)BATCH * HW;  // 131072 rows
  _Float16* visb = (_Float16*)d_ws;                   // [MTOT*512] f16
  _Float16* wqb  = visb + MTOT * DCH;                 // q, then reused for Ww output
  _Float16* kbuf = wqb + MTOT * DCH;                  // [8*32*512] f16
  _Float16* vT   = kbuf + (size_t)BATCH * NLP * DCH;  // [8*512*32] f16
  float* pS    = (float*)(vT + (size_t)BATCH * DCH * NLP);  // [1024*512]
  float* pQ    = pS + (size_t)1024 * DCH;
  float* meanA = pQ + (size_t)1024 * DCH;             // [4096]
  float* invA  = meanA + BATCH * 512;

  _Float16* ao = (_Float16*)d_out;  // attention output staged f16 in d_out low half

  // 1) vis = gelu(x@Wv+bv), qraw = x@Wq+bq (+ fused q stats partials)
  k_gemm_x<<<dim3(8, 1024), 256, 0, stream>>>(x, Wv, bv, Wq, bq, visb, wqb, pS, pQ);
  k_finstats<<<16, 256, 0, stream>>>(pS, pQ, meanA, invA);
  // 2) language K/V projections
  k_kv<<<BATCH * NLP, 256, 0, stream>>>(l, Wk, bk, Wval, bval, kbuf, vT);
  // 3) multi-head attention over 20 tokens
  k_attn<<<dim3(HW / 128, BATCH), 256, 0, stream>>>(wqb, meanA, invA, kbuf, vT, ao);
  // 4) wraw = attn@Ww+bw (f16, reuse q buffer; fused stats partials)
  k_gemm_w<<<dim3(4, 1024), 256, 0, stream>>>(ao, Ww, bw, wqb, pS, pQ);
  k_finstats<<<16, 256, 0, stream>>>(pS, pQ, meanA, invA);
  // 5) out = gelu((vis * inorm(wraw)) @ Wm + bm)
  k_gemm_mm<<<dim3(4, 1024), 256, 0, stream>>>(visb, wqb, meanA, invA, Wm, bm, out);
}
